// DoubleStreamBlock_82686710383290
// MI455X (gfx1250) — matrix operand order repaired and checked by reading, not compiled, not run on hardware
//
#include <hip/hip_runtime.h>
#include <hip/hip_bf16.h>
#include <math.h>

// ---------------------------------------------------------------------------
// CDNA5 (gfx1250) DoubleStreamBlock. All large GEMMs + attention use
// v_wmma_f32_16x16x32_bf16 (wave32). fp32 -> bf16 conversion on the fly.
// B staged transposed in LDS so every WMMA fragment loads as ds_load_b128.
// Staging is software-pipelined: all global loads issued before converts.
// ---------------------------------------------------------------------------

typedef __bf16 v16bf __attribute__((ext_vector_type(16)));
typedef __bf16 v8bf  __attribute__((ext_vector_type(8)));
typedef __bf16 v4bf  __attribute__((ext_vector_type(4)));
typedef float  v8f   __attribute__((ext_vector_type(8)));

#define HIDN 2048
#define QKVW 6144
#define NHEAD 16
#define HDIM 128
#define LTXT 512
#define NIMG 2048
#define LKV  2560   // 512 + 2048
#define MHID 8192

__device__ __forceinline__ v8f wmma_bf16(v16bf a, v16bf b, v8f c) {
    return __builtin_amdgcn_wmma_f32_16x16x32_bf16(false, a, false, b,
                                                   (short)0, c, false, false);
}

__device__ __forceinline__ float gelu_tanh(float x) {
    float x3 = x * x * x;
    return 0.5f * x * (1.0f + tanhf(0.7978845608028654f * (x + 0.044715f * x3)));
}

// Pack a float4 into 4 bf16 lanes of a v4bf.
__device__ __forceinline__ v4bf pack4(float4 v) {
    v4bf p;
    p[0] = (__bf16)v.x; p[1] = (__bf16)v.y;
    p[2] = (__bf16)v.z; p[3] = (__bf16)v.w;
    return p;
}

// ---------------------------------------------------------------------------
// AdaLN embedding: e = silu(vec) @ W + b   (1x2048 @ 2048x12288), img & txt.
// grid (192, 2), block 256. Each block: 64 output cols, 4 K-partials/col.
// ---------------------------------------------------------------------------
__global__ __launch_bounds__(256) void adaln_embed_kernel(
    const float* __restrict__ vec,
    const float* __restrict__ Wimg, const float* __restrict__ bimg,
    const float* __restrict__ Wtxt, const float* __restrict__ btxt,
    float* __restrict__ e_img, float* __restrict__ e_txt)
{
    __shared__ float sv[HIDN];
    __shared__ float red[256];
    const float* W = blockIdx.y ? Wtxt : Wimg;
    const float* bb = blockIdx.y ? btxt : bimg;
    float* e = blockIdx.y ? e_txt : e_img;
    int tid = threadIdx.x;
    for (int i = tid; i < HIDN; i += 256) {
        float x = vec[i];
        sv[i] = x / (1.0f + expf(-x));
    }
    __syncthreads();
    int col = blockIdx.x * 64 + (tid >> 2);
    int part = tid & 3;
    float s = 0.0f;
    int k0 = part * 512;
    for (int k = k0; k < k0 + 512; ++k)
        s += sv[k] * W[(size_t)k * 12288 + col];
    red[tid] = s;
    __syncthreads();
    if (part == 0)
        e[col] = red[tid] + red[tid + 1] + red[tid + 2] + red[tid + 3] + bb[col];
}

// ---------------------------------------------------------------------------
// temperature scalar: scales[0] = hd^-0.5 ; scales[1] = hd^-0.5 / temp
// (s_eff^2 folded into the score scale). block = 64 threads.
// ---------------------------------------------------------------------------
__global__ void temp_kernel(const float* __restrict__ sol_temp,
                            float* __restrict__ scales)
{
    __shared__ float red[64];
    int tid = threadIdx.x;
    red[tid] = sol_temp[tid];
    __syncthreads();
    for (int off = 32; off > 0; off >>= 1) {
        if (tid < off) red[tid] += red[tid + off];
        __syncthreads();
    }
    if (tid == 0) {
        float temp = fmaxf(red[0] / 64.0f, 0.1f);
        const float base = 0.08838834764831845f; // 128^-0.5
        scales[0] = base;
        scales[1] = base / temp;
    }
}

// ---------------------------------------------------------------------------
// mod[h][n] = exp(clip(bilinear(sol_spatial)[n] * mod_w[h] + mod_b[h], -2, 2))
// 8x8 -> 64x32 half-pixel bilinear with edge clamp. 16*2048 threads.
// ---------------------------------------------------------------------------
__global__ void mod_kernel(const float* __restrict__ sol_spatial,
                           const float* __restrict__ mod_w,
                           const float* __restrict__ mod_b,
                           float* __restrict__ modbuf)
{
    int idx = blockIdx.x * blockDim.x + threadIdx.x;
    if (idx >= NHEAD * NIMG) return;
    int h = idx >> 11;
    int n = idx & (NIMG - 1);
    int y = n >> 5, x = n & 31;
    float sy = (y + 0.5f) * 0.125f - 0.5f;
    float sx = (x + 0.5f) * 0.25f - 0.5f;
    int y0 = (int)floorf(sy); float fy = sy - (float)y0;
    int x0 = (int)floorf(sx); float fx = sx - (float)x0;
    int y0c = y0 < 0 ? 0 : (y0 > 7 ? 7 : y0);
    int y1c = (y0 + 1) < 0 ? 0 : ((y0 + 1) > 7 ? 7 : (y0 + 1));
    int x0c = x0 < 0 ? 0 : (x0 > 7 ? 7 : x0);
    int x1c = (x0 + 1) < 0 ? 0 : ((x0 + 1) > 7 ? 7 : (x0 + 1));
    float v00 = sol_spatial[y0c * 8 + x0c], v01 = sol_spatial[y0c * 8 + x1c];
    float v10 = sol_spatial[y1c * 8 + x0c], v11 = sol_spatial[y1c * 8 + x1c];
    float v = v00 * (1.f - fy) * (1.f - fx) + v01 * (1.f - fy) * fx +
              v10 * fy * (1.f - fx) + v11 * fy * fx;
    v = v * mod_w[h] + mod_b[h];
    v = fminf(fmaxf(v, -2.0f), 2.0f);
    modbuf[idx] = expf(v);
}

// ---------------------------------------------------------------------------
// RMS-norm row: y = rms(x)*w*(1+sc) + sh.   grid = rows, block = 256.
// ---------------------------------------------------------------------------
__global__ __launch_bounds__(256) void rmsnorm_kernel(
    const float* __restrict__ X, const float* __restrict__ w,
    const float* __restrict__ scv, const float* __restrict__ shv,
    float* __restrict__ Y)
{
    __shared__ float red[256];
    int row = blockIdx.x, tid = threadIdx.x;
    const float* x = X + (size_t)row * HIDN;
    float s = 0.0f;
    for (int i = tid; i < HIDN; i += 256) { float v = x[i]; s += v * v; }
    red[tid] = s;
    __syncthreads();
    for (int off = 128; off > 0; off >>= 1) {
        if (tid < off) red[tid] += red[tid + off];
        __syncthreads();
    }
    float r = rsqrtf(red[0] * (1.0f / HIDN) + 1e-6f);
    float* y = Y + (size_t)row * HIDN;
    for (int i = tid; i < HIDN; i += 256)
        y[i] = x[i] * r * w[i] * (1.0f + scv[i]) + shv[i];
}

// ---------------------------------------------------------------------------
// RoPE + per-head mod, applied in place to q and k sections of imgQKV.
// One thread per even/odd pair. idx = ((sec*2048 + n)*16 + h)*64 + pair
// ---------------------------------------------------------------------------
__global__ void rope_mod_kernel(float* __restrict__ imgQKV,
                                const float* __restrict__ rope,
                                const float* __restrict__ modbuf)
{
    int idx = blockIdx.x * blockDim.x + threadIdx.x;
    if (idx >= NIMG * NHEAD * 64 * 2) return;
    int pair = idx & 63;
    int h = (idx >> 6) & 15;
    int n = (idx >> 10) & (NIMG - 1);
    int sec = idx >> 21; // 0 = q, 1 = k
    size_t base = (size_t)n * QKVW + sec * HIDN + h * HDIM + pair * 2;
    float x0 = imgQKV[base], x1 = imgQKV[base + 1];
    float c = rope[n * HDIM + pair * 2];
    float s = rope[n * HDIM + pair * 2 + 1];
    float m = modbuf[h * NIMG + n];
    imgQKV[base]     = (x0 * c - x1 * s) * m;
    imgQKV[base + 1] = (x1 * c + x0 * s) * m;
}

// ---------------------------------------------------------------------------
// Generic bf16-WMMA GEMM: C = epilogue(A @ B).
// A: MxK fp32 row-major, B: KxN fp32 row-major. WG tile 128x64, 8 waves,
// each wave -> 64(M)x16(N), K staged in LDS as bf16 in chunks of 32.
// As is [m][k]; B is staged TRANSPOSED as Bst[n][k] so both A and B
// fragments are contiguous 16B runs -> ds_load_b128.
// Staging: all global b128 loads issued to registers first, then cvt+store.
// mode: 0 = plain ; 1 = +bias, gelu ; 2 = res + gate[n]*acc ;
//       3 = res + gate[n]*(acc + bias)
// Requires M%128==0, N%64==0, K%32==0 (true for all uses here).
// ---------------------------------------------------------------------------
__global__ __launch_bounds__(256) void gemm_wmma_kernel(
    const float* __restrict__ A, const float* __restrict__ Bm,
    const float* __restrict__ bias, const float* __restrict__ res,
    const float* __restrict__ gate, float* __restrict__ C,
    int M, int N, int K, int mode)
{
    __shared__ __bf16 As[128][32];   // [m][k]
    __shared__ __bf16 Bst[64][40];   // [n][k], +8 pad
    const int tid  = threadIdx.x;
    const int wave = tid >> 5;
    const int lane = tid & 31;
    const int half = lane >> 4;
    const int l16  = lane & 15;
    const int wn = wave & 3;   // N sub-tile *16
    const int wm = wave >> 2;  // M strip *64
    const int bm = blockIdx.y * 128;
    const int bn = blockIdx.x * 64;

    // Per-thread staging coordinates (compile-time trip counts).
    const int ar[4] = { (tid) >> 3, (tid + 256) >> 3, (tid + 512) >> 3, (tid + 768) >> 3 };
    const int ac = (tid & 7) * 4;
    const int br[2] = { tid >> 4, (tid + 256) >> 4 };
    const int bc = (tid & 15) * 4;

    v8f acc[4];
    for (int s = 0; s < 4; ++s)
        for (int r = 0; r < 8; ++r) acc[s][r] = 0.0f;

    for (int k0 = 0; k0 < K; k0 += 32) {
        // Phase 1: issue all global loads (stay in flight together).
        float4 va[4], vb[2];
#pragma unroll
        for (int j = 0; j < 4; ++j)
            va[j] = *(const float4*)(A + (size_t)(bm + ar[j]) * K + k0 + ac);
#pragma unroll
        for (int j = 0; j < 2; ++j)
            vb[j] = *(const float4*)(Bm + (size_t)(k0 + br[j]) * N + bn + bc);
        __syncthreads();
        // Phase 2: convert + LDS stores.
#pragma unroll
        for (int j = 0; j < 4; ++j)
            *(v4bf*)&As[ar[j]][ac] = pack4(va[j]);
#pragma unroll
        for (int j = 0; j < 2; ++j) {
            Bst[bc + 0][br[j]] = (__bf16)vb[j].x;
            Bst[bc + 1][br[j]] = (__bf16)vb[j].y;
            Bst[bc + 2][br[j]] = (__bf16)vb[j].z;
            Bst[bc + 3][br[j]] = (__bf16)vb[j].w;
        }
        __syncthreads();
        // B fragment: k = half*16 + i, n = wn*16+l16 -> contiguous in Bst row
        v16bf bf;
        {
            const __bf16* bp = &Bst[wn * 16 + l16][half * 16];
            ((v8bf*)&bf)[0] = *(const v8bf*)bp;
            ((v8bf*)&bf)[1] = *(const v8bf*)(bp + 8);
        }
#pragma unroll
        for (int sub = 0; sub < 4; ++sub) {
            int r0 = wm * 64 + sub * 16 + l16;
            v16bf af;
            const __bf16* ap = &As[r0][half * 16];
            ((v8bf*)&af)[0] = *(const v8bf*)ap;          // k = half*16 .. +7
            ((v8bf*)&af)[1] = *(const v8bf*)(ap + 8);    // k = half*16+8 .. +7
            acc[sub] = wmma_bf16(af, bf, acc[sub]);
        }
    }

    int gn = bn + wn * 16 + l16;
    float bval = (mode == 1 || mode == 3) ? bias[gn] : 0.0f;
    float gval = (mode >= 2) ? gate[gn] : 0.0f;
    for (int sub = 0; sub < 4; ++sub) {
        for (int r = 0; r < 8; ++r) {
            int gm = bm + wm * 64 + sub * 16 + half * 8 + r;
            float v = acc[sub][r] + bval;
            if (mode == 1) v = gelu_tanh(v);
            if (mode >= 2) v = res[(size_t)gm * N + gn] + gval * v;
            C[(size_t)gm * N + gn] = v;
        }
    }
}

// ---------------------------------------------------------------------------
// Attention, two-pass with full score row-block in LDS (16 q rows x 2560 keys
// = 160KB, within the 320KB WGP budget). grid = (Lq/16, NHEAD), block 256.
// Q from Qbuf (rows x 6144, q at col h*128+d); K at col 2048+..., V at 4096+...
// keys 0..511 from txtKV rows, 512..2559 from imgKV rows.
// Out: (Lq, 2048) at col h*128+d (already transposed/merged layout).
// ---------------------------------------------------------------------------
__global__ __launch_bounds__(256) void attn_wmma_kernel(
    const float* __restrict__ Qbuf, const float* __restrict__ txtKV,
    const float* __restrict__ imgKV, const float* __restrict__ scales,
    int scale_idx, float* __restrict__ Out, int Lq)
{
    extern __shared__ float smem[];
    float* S     = smem;               // [16][2560]
    float* red   = smem + 16 * LKV;    // [16][16]
    float* rstat = red + 256;          // [16]

    const int tid  = threadIdx.x;
    const int wave = tid >> 5;
    const int lane = tid & 31;
    const int half = lane >> 4;
    const int l16  = lane & 15;
    const int q0   = blockIdx.x * 16;
    const int h    = blockIdx.y;
    const int qcol = h * HDIM;
    const float sc = scales[scale_idx];

    // Hoisted Q fragments (A-matrix 16x32, 4 chunks cover d=128), float4 loads
    v16bf qf[4];
#pragma unroll
    for (int kc = 0; kc < 4; ++kc) {
        const float* qrow = Qbuf + (size_t)(q0 + l16) * QKVW + qcol + kc * 32 + half * 8;
        ((v4bf*)&qf[kc])[0] = pack4(*(const float4*)(qrow));
        ((v4bf*)&qf[kc])[1] = pack4(*(const float4*)(qrow + 4));
        ((v4bf*)&qf[kc])[2] = pack4(*(const float4*)(qrow + 16));
        ((v4bf*)&qf[kc])[3] = pack4(*(const float4*)(qrow + 20));
    }

    // Pass 1: S = sc * Q K^T
    for (int kt = wave; kt < LKV / 16; kt += 8) {
        int key0 = kt * 16;
        v8f acc;
        for (int r = 0; r < 8; ++r) acc[r] = 0.0f;
        int key = key0 + l16; // this lane's key column
        const float* krow = (key < LTXT) ? (txtKV + (size_t)key * QKVW)
                                         : (imgKV + (size_t)(key - LTXT) * QKVW);
#pragma unroll
        for (int kc = 0; kc < 4; ++kc) {
            const float* kp = krow + HIDN + qcol + kc * 32 + half * 8;
            v16bf bf;
            ((v4bf*)&bf)[0] = pack4(*(const float4*)(kp));
            ((v4bf*)&bf)[1] = pack4(*(const float4*)(kp + 4));
            ((v4bf*)&bf)[2] = pack4(*(const float4*)(kp + 16));
            ((v4bf*)&bf)[3] = pack4(*(const float4*)(kp + 20));
            acc = wmma_bf16(qf[kc], bf, acc);
        }
        for (int r = 0; r < 8; ++r)
            S[(half * 8 + r) * LKV + key0 + l16] = acc[r] * sc;
    }
    __syncthreads();

    // Softmax over 2560 per row; 16 threads per row.
    {
        int row = tid >> 4, sub = tid & 15;
        float m = -1e30f;
        for (int c = sub; c < LKV; c += 16) m = fmaxf(m, S[row * LKV + c]);
        red[row * 16 + sub] = m;
        __syncthreads();
        if (sub == 0) {
            float mm = red[row * 16];
            for (int i = 1; i < 16; ++i) mm = fmaxf(mm, red[row * 16 + i]);
            rstat[row] = mm;
        }
        __syncthreads();
        float rmax = rstat[row];
        float ssum = 0.0f;
        for (int c = sub; c < LKV; c += 16) {
            float e = expf(S[row * LKV + c] - rmax);
            S[row * LKV + c] = e;
            ssum += e;
        }
        red[row * 16 + sub] = ssum;
        __syncthreads();
        if (sub == 0) {
            float t = 0.0f;
            for (int i = 0; i < 16; ++i) t += red[row * 16 + i];
            rstat[row] = 1.0f / t;
        }
        __syncthreads();
        float rinv = rstat[row];
        for (int c = sub; c < LKV; c += 16) S[row * LKV + c] *= rinv;
        __syncthreads();
    }

    // Pass 2: O = P V. Each wave owns one 16-wide d sub-tile (8*16 = 128).
    {
        int n0 = wave * 16;
        v8f acc;
        for (int r = 0; r < 8; ++r) acc[r] = 0.0f;
        for (int kc = 0; kc < LKV / 32; ++kc) {
            const float* sp = S + l16 * LKV + kc * 32 + half * 8;
            v16bf af;
            ((v4bf*)&af)[0] = pack4(*(const float4*)(sp));
            ((v4bf*)&af)[1] = pack4(*(const float4*)(sp + 4));
            ((v4bf*)&af)[2] = pack4(*(const float4*)(sp + 16));
            ((v4bf*)&af)[3] = pack4(*(const float4*)(sp + 20));
            v16bf bf;
#pragma unroll
            for (int i = 0; i < 16; ++i) {
                int key = kc * 32 + (i < 8 ? half * 8 + i : 16 + half * 8 + (i - 8));
                const float* vrow = (key < LTXT) ? (txtKV + (size_t)key * QKVW)
                                                 : (imgKV + (size_t)(key - LTXT) * QKVW);
                bf[i] = (__bf16)vrow[2 * HIDN + qcol + n0 + l16];
            }
            acc = wmma_bf16(af, bf, acc);
        }
        for (int r = 0; r < 8; ++r)
            Out[(size_t)(q0 + half * 8 + r) * HIDN + qcol + n0 + l16] = acc[r];
    }
}

// ---------------------------------------------------------------------------
// Host-side launch
// ---------------------------------------------------------------------------
extern "C" void kernel_launch(void* const* d_in, const int* in_sizes, int n_in,
                              void* d_out, int out_size, void* d_ws, size_t ws_size,
                              hipStream_t stream) {
    const float* txt        = (const float*)d_in[0];
    const float* img        = (const float*)d_in[1];
    const float* vec        = (const float*)d_in[2];
    const float* rope       = (const float*)d_in[3];
    const float* sol_temp   = (const float*)d_in[4];
    const float* sol_sp     = (const float*)d_in[5];
    const float* ada_img_w  = (const float*)d_in[6];
    const float* ada_img_b  = (const float*)d_in[7];
    const float* ada_img_nw = (const float*)d_in[8];
    const float* ada_txt_w  = (const float*)d_in[9];
    const float* ada_txt_b  = (const float*)d_in[10];
    const float* ada_txt_nw = (const float*)d_in[11];
    const float* txt_qkv_w  = (const float*)d_in[12];
    const float* img_qkv_w  = (const float*)d_in[13];
    const float* txt_out_w  = (const float*)d_in[14];
    const float* img_out_w  = (const float*)d_in[15];
    const float* mod_w      = (const float*)d_in[16];
    const float* mod_b      = (const float*)d_in[17];
    const float* img_n2_w   = (const float*)d_in[18];
    const float* txt_n2_w   = (const float*)d_in[19];
    const float* img_fc1_w  = (const float*)d_in[20];
    const float* img_fc1_b  = (const float*)d_in[21];
    const float* img_fc2_w  = (const float*)d_in[22];
    const float* img_fc2_b  = (const float*)d_in[23];
    const float* txt_fc1_w  = (const float*)d_in[24];
    const float* txt_fc1_b  = (const float*)d_in[25];
    const float* txt_fc2_w  = (const float*)d_in[26];
    const float* txt_fc2_b  = (const float*)d_in[27];

    float* ws = (float*)d_ws;
    size_t off = 0;
    float* e_img  = ws + off; off += 12288;
    float* e_txt  = ws + off; off += 12288;
    float* modbuf = ws + off; off += (size_t)NHEAD * NIMG;
    float* scales = ws + off; off += 64;
    float* imgbig = ws + off; off += (size_t)NIMG * MHID;  // img QKV, then img MLP hidden
    float* txtbig = ws + off; off += (size_t)LTXT * MHID;  // txt QKV, then txt MLP hidden
    float* imgn   = ws + off; off += (size_t)NIMG * HIDN;  // img_n, then img_in2
    float* txtn   = ws + off; off += (size_t)LTXT * HIDN;  // txt_n, then txt_in2
    float* attn_t = ws + off; off += (size_t)LTXT * HIDN;
    float* attn_i = ws + off; off += (size_t)NIMG * HIDN;

    float* out_txt = (float*)d_out;                      // (512, 2048)
    float* out_img = out_txt + (size_t)LTXT * HIDN;      // (2048, 2048)

    const size_t attn_smem = (16 * (size_t)LKV + 256 + 16) * sizeof(float);

    // 1) AdaLN embeddings, temperature, mod table
    adaln_embed_kernel<<<dim3(192, 2), 256, 0, stream>>>(
        vec, ada_img_w, ada_img_b, ada_txt_w, ada_txt_b, e_img, e_txt);
    temp_kernel<<<1, 64, 0, stream>>>(sol_temp, scales);
    mod_kernel<<<(NHEAD * NIMG + 255) / 256, 256, 0, stream>>>(
        sol_sp, mod_w, mod_b, modbuf);

    // 2) Pre-attention RMS norms (e layout: sh_m|sc_m|g_m|sh_p|sc_p|g_p)
    rmsnorm_kernel<<<NIMG, 256, 0, stream>>>(img, ada_img_nw,
                                             e_img + 2048, e_img, imgn);
    rmsnorm_kernel<<<LTXT, 256, 0, stream>>>(txt, ada_txt_nw,
                                             e_txt + 2048, e_txt, txtn);

    // 3) QKV projections
    gemm_wmma_kernel<<<dim3(QKVW / 64, NIMG / 128), 256, 0, stream>>>(
        imgn, img_qkv_w, nullptr, nullptr, nullptr, imgbig, NIMG, QKVW, HIDN, 0);
    gemm_wmma_kernel<<<dim3(QKVW / 64, LTXT / 128), 256, 0, stream>>>(
        txtn, txt_qkv_w, nullptr, nullptr, nullptr, txtbig, LTXT, QKVW, HIDN, 0);

    // 4) RoPE + mod on img q/k (in place)
    rope_mod_kernel<<<(NIMG * NHEAD * 64 * 2) / 256, 256, 0, stream>>>(
        imgbig, rope, modbuf);

    // 5) Attention (txt queries, then img queries w/ temp-folded scale)
    attn_wmma_kernel<<<dim3(LTXT / 16, NHEAD), 256, attn_smem, stream>>>(
        txtbig, txtbig, imgbig, scales, 0, attn_t, LTXT);
    attn_wmma_kernel<<<dim3(NIMG / 16, NHEAD), 256, attn_smem, stream>>>(
        imgbig, txtbig, imgbig, scales, 1, attn_i, NIMG);

    // 6) Output projections, fused gated residual -> d_out
    gemm_wmma_kernel<<<dim3(HIDN / 64, LTXT / 128), 256, 0, stream>>>(
        attn_t, txt_out_w, nullptr, txt, e_txt + 4096, out_txt, LTXT, HIDN, HIDN, 2);
    gemm_wmma_kernel<<<dim3(HIDN / 64, NIMG / 128), 256, 0, stream>>>(
        attn_i, img_out_w, nullptr, img, e_img + 4096, out_img, NIMG, HIDN, HIDN, 2);

    // 7) Post-attention RMS norms (sc_p / sh_p)
    rmsnorm_kernel<<<LTXT, 256, 0, stream>>>(out_txt, txt_n2_w,
                                             e_txt + 8192, e_txt + 6144, txtn);
    rmsnorm_kernel<<<NIMG, 256, 0, stream>>>(out_img, img_n2_w,
                                             e_img + 8192, e_img + 6144, imgn);

    // 8) MLP fc1 + GELU (reuse big buffers; QKV data is dead now)
    gemm_wmma_kernel<<<dim3(MHID / 64, LTXT / 128), 256, 0, stream>>>(
        txtn, txt_fc1_w, txt_fc1_b, nullptr, nullptr, txtbig, LTXT, MHID, HIDN, 1);
    gemm_wmma_kernel<<<dim3(MHID / 64, NIMG / 128), 256, 0, stream>>>(
        imgn, img_fc1_w, img_fc1_b, nullptr, nullptr, imgbig, NIMG, MHID, HIDN, 1);

    // 9) MLP fc2, fused bias + gate (g_p) + in-place residual into d_out
    gemm_wmma_kernel<<<dim3(HIDN / 64, LTXT / 128), 256, 0, stream>>>(
        txtbig, txt_fc2_w, txt_fc2_b, out_txt, e_txt + 10240, out_txt,
        LTXT, HIDN, MHID, 3);
    gemm_wmma_kernel<<<dim3(HIDN / 64, NIMG / 128), 256, 0, stream>>>(
        imgbig, img_fc2_w, img_fc2_b, out_img, e_img + 10240, out_img,
        NIMG, HIDN, MHID, 3);

    (void)in_sizes; (void)n_in; (void)out_size; (void)ws_size;
}
